// VideokMaXTransformerDecoder_57054345560107
// MI455X (gfx1250) — matrix operand order repaired and checked by reading, not compiled, not run on hardware
//
#include <hip/hip_runtime.h>
#include <hip/hip_bf16.h>
#include <math.h>

// ---------------- problem sizes ----------------
#define NB   8
#define CP   2048
#define HWD  4096
#define CQ   256
#define LQ   128
#define NHD  8
#define DV   32
#define KD   128
#define VDEP 256
#define BOT  256
#define FFN  2048
#define MT   4224   // LQ + HWD

#define DI __device__ __forceinline__

typedef __attribute__((ext_vector_type(16))) __bf16 v16bf;
typedef __attribute__((ext_vector_type(8)))  float  v8f;

union Frag16 { v16bf v; uint4 q[2]; };
union Pack8  { uint4 q; unsigned short u[8]; };

DI unsigned short f2bf(float x) {
  union { __hip_bfloat16 b; unsigned short u; } cv;
  cv.b = __float2bfloat16(x);
  return cv.u;
}
DI float geluf(float x) { return 0.5f * x * (1.0f + erff(x * 0.70710678118654752f)); }

// ---------------- Tensor Data Mover (CDNA5) support ----------------
#if defined(__HIP_DEVICE_COMPILE__) && \
    __has_builtin(__builtin_amdgcn_tensor_load_to_lds) && \
    __has_builtin(__builtin_amdgcn_s_wait_tensorcnt)
#define USE_TDM 1
#else
#define USE_TDM 0
#endif

#if USE_TDM
typedef __attribute__((ext_vector_type(4))) unsigned int u32x4;
typedef __attribute__((ext_vector_type(8))) int i32x8;
typedef __attribute__((ext_vector_type(4))) int i32x4;

// Issue a 2D TDM load of a [d1_rows x d0_elems] tile of 2-byte elements
// (row stride = stride_elems) from global memory into LDS at lds_byte_off.
// D# layout per CDNA5 ISA §8.3-8.5: group0 = {flags, lds_addr, global_addr, type=2},
// group1 = {data_size, tensor dims, tile dims, dim0 stride}; groups 2/3 zero (<=2D).
DI void tdm_load_2d_b16(unsigned lds_byte_off, const void* gptr,
                        unsigned d0_elems, unsigned d1_rows, unsigned stride_elems) {
  unsigned long long ga = (unsigned long long)gptr;
  u32x4 g0;
  g0.x = 1u;                                                // count=1, user mode
  g0.y = lds_byte_off;                                      // lds_addr (bytes)
  g0.z = (unsigned)ga;                                      // global_addr[31:0]
  g0.w = (unsigned)((ga >> 32) & 0x01FFFFFFu) | (2u << 30); // addr[56:32] | type=2
  i32x8 g1;
  g1[0] = (int)(1u << 16);                                   // data_size = 1 -> 2 bytes
  g1[1] = (int)((d0_elems & 0xFFFFu) << 16);                 // tensor_dim0[15:0]
  g1[2] = (int)(((d0_elems >> 16) & 0xFFFFu) |
                ((d1_rows & 0xFFFFu) << 16));                // tensor_dim0 hi | tensor_dim1 lo
  g1[3] = (int)(((d1_rows >> 16) & 0xFFFFu) |
                ((d0_elems & 0xFFFFu) << 16));               // tensor_dim1 hi | tile_dim0
  g1[4] = (int)(d1_rows & 0xFFFFu);                          // tile_dim1 (tile_dim2 = 0)
  g1[5] = (int)stride_elems;                                 // tensor_dim0_stride[31:0]
  g1[6] = 0;
  g1[7] = 0;
  i32x4 gz = {0, 0, 0, 0};
#if defined(__clang_major__) && __clang_major__ >= 23
  i32x8 gz8 = {0, 0, 0, 0, 0, 0, 0, 0};
  __builtin_amdgcn_tensor_load_to_lds(g0, g1, gz, gz, gz8, 0);
#else
  __builtin_amdgcn_tensor_load_to_lds(g0, g1, gz, gz, 0);
#endif
}
#endif

// B-source modes and epilogue modes
constexpr int B_CF_GELU = 0;  // channel-first fp32, gelu on load, transpose into LDS
constexpr int B_CF      = 1;  // channel-first fp32, transpose into LDS
constexpr int B_CL      = 2;  // channel-last bf16, direct copy into LDS

constexpr int EPI_GELU_CL = 0;  // y=gelu(bn(x)) -> bf16 channel-last [n][m][O]
constexpr int EPI_PKV     = 1;  // y=bn(x) -> kT / v scatter (pixel part, m+=128)
constexpr int EPI_QKV     = 2;  // y=bn(x) -> qT / kT / v scatter (query part)
constexpr int EPI_QF      = 3;  // y=gelu(bn(x)+resid) -> bf16 CL + fp32 CF
constexpr int EPI_OUT     = 4;  // y=gelu(bn(x)+resid) -> fp32 CF (d_out)

struct GemmArgs {
  const unsigned short* A;   // bf16 weights [O][K]
  const void* B;             // per BMODE
  const float* s;            // BN scale [O]
  const float* bias;         // BN bias  [O]
  int O, K, M;
  long long bstride;         // B elements per batch
  unsigned short* out_cl;    // channel-last bf16 [n][M][O]
  unsigned short* kT;        // [n][h][MT][32] (dk padded to 32, zero pad)
  unsigned short* vB;        // [n][h][32][MT]
  unsigned short* qT;        // [n][h][LQ][32] (dk padded to 32, zero pad)
  int m_off;                 // m offset into kT/vB (128 for pixel tokens)
  const float* resid;        // fp32 channel-first [n][CQ][LQ]
  float* out_f32;            // fp32 channel-first [n][CQ][LQ]
};

// ---------------- generic WMMA GEMM: C[o][m] = sum_k A[o][k] * B[k][m] ----------------
// block = 256 threads = 8 waves; block tile 128(O) x 128(M); K step 32.
// wave (wo,wm): wo in {0,1} over O halves (64 rows), wm in {0..3} over M quarters (32 cols).
template<int BMODE, int EPI>
__global__ __launch_bounds__(256) void gemm_wmma(GemmArgs a) {
  __shared__ __align__(16) unsigned short lA[128 * 32];
  __shared__ __align__(16) unsigned short lB[128 * 32];

  const int tid  = threadIdx.x;
  const int lane = tid & 31;
  const int wv   = tid >> 5;
  const int wo   = wv & 1;
  const int wm   = wv >> 1;
  const int n      = blockIdx.z;
  const int o_base = blockIdx.y * 128;
  const int m_base = blockIdx.x * 128;

  v8f acc[4][2];
#pragma unroll
  for (int i = 0; i < 4; i++) { acc[i][0] = {}; acc[i][1] = {}; }

  for (int k0 = 0; k0 < a.K; k0 += 32) {
    // ---- A tile: 128 rows x 32 halves (row-major, K contiguous) ----
#pragma unroll
    for (int i = 0; i < 2; i++) {
      int idx = tid + i * 256;
      int r = idx >> 2, ch = idx & 3;
      const uint4* src = reinterpret_cast<const uint4*>(a.A + (size_t)(o_base + r) * a.K + k0) + ch;
      reinterpret_cast<uint4*>(lA)[r * 4 + ch] = *src;
    }
    if (k0 + 32 < a.K) {
      __builtin_prefetch(a.A + (size_t)(o_base + (tid >> 1)) * a.K + k0 + 32, 0, 0);
    }
    // ---- B tile -> lB[m][k], 128 rows(m) x 32 halves(k) ----
    if constexpr (BMODE == B_CL) {
      const unsigned short* Bg = (const unsigned short*)a.B + (size_t)n * a.bstride;
#pragma unroll
      for (int i = 0; i < 2; i++) {
        int idx = tid + i * 256;
        int r = idx >> 2, ch = idx & 3;
        const uint4* src = reinterpret_cast<const uint4*>(Bg + (size_t)(m_base + r) * a.K + k0) + ch;
        reinterpret_cast<uint4*>(lB)[r * 4 + ch] = *src;
      }
    } else {
      // channel-first fp32 [K][M]: coalesced read along M of two adjacent K rows,
      // pack 2 bf16 per b32 LDS store (transpose+convert into LDS)
      const float* Bg = (const float*)a.B + (size_t)n * a.bstride;
      int kp = (tid >> 4) * 2;     // even k: 0,2,...,30
      int m0 = (tid & 15) * 8;     // 8 m per thread
      const float4* s0 = reinterpret_cast<const float4*>(Bg + (size_t)(k0 + kp) * a.M + m_base + m0);
      const float4* s1 = reinterpret_cast<const float4*>(Bg + (size_t)(k0 + kp + 1) * a.M + m_base + m0);
      float4 a0 = s0[0], a1 = s0[1];
      float4 b0 = s1[0], b1 = s1[1];
      float r0[8] = { a0.x, a0.y, a0.z, a0.w, a1.x, a1.y, a1.z, a1.w };
      float r1[8] = { b0.x, b0.y, b0.z, b0.w, b1.x, b1.y, b1.z, b1.w };
#pragma unroll
      for (int t = 0; t < 8; t++) {
        float x0 = r0[t], x1 = r1[t];
        if constexpr (BMODE == B_CF_GELU) { x0 = geluf(x0); x1 = geluf(x1); }
        unsigned pk = (unsigned)f2bf(x0) | ((unsigned)f2bf(x1) << 16);
        *reinterpret_cast<unsigned*>(&lB[(m0 + t) * 32 + kp]) = pk;
      }
    }
    __syncthreads();

    // ---- fragments ----
    Frag16 af[4], bq[2];
    const int c0 = (lane < 16) ? 0 : 16;  // A: lanes 0-15 K=0..15, lanes 16-31 K=16..31
#pragma unroll
    for (int oi = 0; oi < 4; oi++) {
      int row = wo * 64 + oi * 16 + (lane & 15);
      const uint4* p = reinterpret_cast<const uint4*>(lA + row * 32);
      af[oi].q[0] = p[c0 >> 3];
      af[oi].q[1] = p[(c0 >> 3) + 1];
    }
    const int kb = (lane < 16) ? 0 : 16;  // B: lanes 0-15 K=0..15, lanes 16-31 K=16..31
#pragma unroll
    for (int mi = 0; mi < 2; mi++) {
      int row = wm * 32 + mi * 16 + (lane & 15);
      const uint4* p = reinterpret_cast<const uint4*>(lB + row * 32 + kb);
      bq[mi].q[0] = p[0];
      bq[mi].q[1] = p[1];
    }
#pragma unroll
    for (int oi = 0; oi < 4; oi++)
#pragma unroll
      for (int mi = 0; mi < 2; mi++)
        acc[oi][mi] = __builtin_amdgcn_wmma_f32_16x16x32_bf16(
            false, af[oi].v, false, bq[mi].v, (short)0, acc[oi][mi], false, false);
    __syncthreads();
  }

  // ---- epilogue: C element (VGPR r, lane): row = r + 8*(lane>>4), col = lane&15 ----
  const int rh  = lane >> 4;
  const int col = lane & 15;
#pragma unroll
  for (int oi = 0; oi < 4; oi++) {
#pragma unroll
    for (int mi = 0; mi < 2; mi++) {
      const int o0 = o_base + wo * 64 + oi * 16 + rh * 8;  // 8 consecutive output channels
      const int m  = m_base + wm * 32 + mi * 16 + col;
      float y[8];
#pragma unroll
      for (int r = 0; r < 8; r++) y[r] = acc[oi][mi][r] * a.s[o0 + r] + a.bias[o0 + r];

      if constexpr (EPI == EPI_GELU_CL) {
        Pack8 pk;
#pragma unroll
        for (int r = 0; r < 8; r++) pk.u[r] = f2bf(geluf(y[r]));
        *reinterpret_cast<uint4*>(a.out_cl + ((size_t)n * a.M + m) * a.O + o0) = pk.q;
      } else if constexpr (EPI == EPI_PKV) {
        if (o0 < KD) {  // pixel_k -> kT[n][h][128+m][32] at d
          int h = o0 >> 4, d0 = o0 & 15;
          Pack8 pk;
#pragma unroll
          for (int r = 0; r < 8; r++) pk.u[r] = f2bf(y[r]);
          *reinterpret_cast<uint4*>(a.kT + ((size_t)(n * NHD + h) * MT + (m + a.m_off)) * 32 + d0) = pk.q;
        } else {        // pixel_v -> v[n][h][d][128+m]
          int o2 = o0 - KD;
          int h = o2 >> 5, d0 = o2 & 31;
#pragma unroll
          for (int r = 0; r < 8; r++)
            a.vB[((size_t)(n * NHD + h) * DV + d0 + r) * MT + (m + a.m_off)] = f2bf(y[r]);
        }
      } else if constexpr (EPI == EPI_QKV) {
        if (o0 < KD) {            // q -> qT[n][h][l][32]
          int h = o0 >> 4, d0 = o0 & 15;
          Pack8 pk;
#pragma unroll
          for (int r = 0; r < 8; r++) pk.u[r] = f2bf(y[r]);
          *reinterpret_cast<uint4*>(a.qT + ((size_t)(n * NHD + h) * LQ + m) * 32 + d0) = pk.q;
        } else if (o0 < 2 * KD) { // qk -> kT[n][h][m][32]
          int o2 = o0 - KD;
          int h = o2 >> 4, d0 = o2 & 15;
          Pack8 pk;
#pragma unroll
          for (int r = 0; r < 8; r++) pk.u[r] = f2bf(y[r]);
          *reinterpret_cast<uint4*>(a.kT + ((size_t)(n * NHD + h) * MT + m) * 32 + d0) = pk.q;
        } else {                  // qv -> v[n][h][d][m]
          int o2 = o0 - 2 * KD;
          int h = o2 >> 5, d0 = o2 & 31;
#pragma unroll
          for (int r = 0; r < 8; r++)
            a.vB[((size_t)(n * NHD + h) * DV + d0 + r) * MT + m] = f2bf(y[r]);
        }
      } else if constexpr (EPI == EPI_QF) {
        Pack8 pk;
#pragma unroll
        for (int r = 0; r < 8; r++) {
          int o = o0 + r;
          float v = y[r] + a.resid[((size_t)n * CQ + o) * LQ + m];
          float g = geluf(v);
          pk.u[r] = f2bf(g);
          a.out_f32[((size_t)n * CQ + o) * LQ + m] = g;
        }
        *reinterpret_cast<uint4*>(a.out_cl + ((size_t)n * LQ + m) * CQ + o0) = pk.q;
      } else if constexpr (EPI == EPI_OUT) {
#pragma unroll
        for (int r = 0; r < 8; r++) {
          int o = o0 + r;
          float v = y[r] + a.resid[((size_t)n * CQ + o) * LQ + m];
          a.out_f32[((size_t)n * CQ + o) * LQ + m] = geluf(v);
        }
      }
    }
  }
}

// ---------------- flash-attention with WMMA + TDM staging ----------------
// grid = N*NH blocks; block = 256 = 8 waves; wave w owns query rows [w*16, w*16+16).
// Per 32-token chunk: wave 0 DMAs the k chunk (32x32 bf16) and v chunk (32 rows,
// stride MT) into LDS via tensor_load_to_lds (TENSORcnt), then all waves build
// fragments from LDS. sim = q^T k (dk=16 zero-padded to K=32); online softmax;
// ret += P v^T via WMMA; epilogue = bn + gelu -> retT[n][l][VDEP] bf16.
__global__ __launch_bounds__(256) void attn_wmma(
    const unsigned short* __restrict__ qT, const unsigned short* __restrict__ kT,
    const unsigned short* __restrict__ vB,
    const float* __restrict__ s_sim, const float* __restrict__ b_sim,
    const float* __restrict__ s_ret, const float* __restrict__ b_ret,
    unsigned short* __restrict__ retT) {
  // single LDS arena so TDM byte offsets are known (group segment starts at 0):
  //   [0,8192)      lP: 8 waves x 16x32 bf16 (P transpose buffers)
  //   [8192,10240)  lK: 32(m) x 32(half) bf16 chunk
  //   [10240,12288) lV: 32(d) x 32(half) bf16 chunk
  __shared__ __align__(16) unsigned short smem[8 * 16 * 32 + 32 * 32 + 32 * 32];
  unsigned short* lP = smem;
  unsigned short* lK = smem + 4096;
  unsigned short* lV = smem + 4096 + 1024;

  const int tid = threadIdx.x, lane = tid & 31, wv = tid >> 5;
  const int n = blockIdx.x >> 3, h = blockIdx.x & 7;
  const int lb = wv * 16;

  Frag16 aq;
  {
    const unsigned short* qrow = qT + ((size_t)(n * NHD + h) * LQ + lb + (lane & 15)) * 32;
    const uint4* p = reinterpret_cast<const uint4*>(qrow);
    int c0 = (lane < 16) ? 0 : 16;
    aq.q[0] = p[c0 >> 3];
    aq.q[1] = p[(c0 >> 3) + 1];
  }
  const float ss = s_sim[h], sb = b_sim[h];
  const unsigned short* kTh = kT + (size_t)(n * NHD + h) * MT * 32;
  const unsigned short* vh  = vB + (size_t)(n * NHD + h) * DV * MT;

  float mrow[8], lrow[8];
#pragma unroll
  for (int r = 0; r < 8; r++) { mrow[r] = -1e30f; lrow[r] = 0.0f; }
  v8f acc0 = {}, acc1 = {};

  const int kb = (lane < 16) ? 0 : 16;
  for (int mc = 0; mc < MT; mc += 32) {
    __syncthreads();  // previous iteration's LDS reads complete
#if USE_TDM
    if (wv == 0) {
      tdm_load_2d_b16(8192u,  kTh + (size_t)mc * 32, 32u, 32u, 32u);
      tdm_load_2d_b16(10240u, vh + mc,               32u, 32u, (unsigned)MT);
      __builtin_amdgcn_s_wait_tensorcnt(0);
    }
    __syncthreads();  // lK / lV visible to all waves
    Frag16 bk0, bk1;
    {
      const uint4* p0 = reinterpret_cast<const uint4*>(lK + (size_t)(lane & 15) * 32 + kb);
      bk0.q[0] = p0[0]; bk0.q[1] = p0[1];
      const uint4* p1 = reinterpret_cast<const uint4*>(lK + (size_t)(16 + (lane & 15)) * 32 + kb);
      bk1.q[0] = p1[0]; bk1.q[1] = p1[1];
    }
#else
    Frag16 bk0, bk1;
    {
      const uint4* p0 = reinterpret_cast<const uint4*>(kTh + (size_t)(mc + (lane & 15)) * 32 + kb);
      bk0.q[0] = p0[0]; bk0.q[1] = p0[1];
      const uint4* p1 = reinterpret_cast<const uint4*>(kTh + (size_t)(mc + 16 + (lane & 15)) * 32 + kb);
      bk1.q[0] = p1[0]; bk1.q[1] = p1[1];
    }
#endif
    v8f z = {};
    v8f s0 = __builtin_amdgcn_wmma_f32_16x16x32_bf16(false, aq.v, false, bk0.v, (short)0, z, false, false);
    v8f s1 = __builtin_amdgcn_wmma_f32_16x16x32_bf16(false, aq.v, false, bk1.v, (short)0, z, false, false);

    // online softmax: row lives across the 16 lanes of a half-wave
    float p0v[8], p1v[8], corr[8];
#pragma unroll
    for (int r = 0; r < 8; r++) {
      float x0 = s0[r] * ss + sb;
      float x1 = s1[r] * ss + sb;
      float t = fmaxf(x0, x1);
#pragma unroll
      for (int msk = 1; msk < 16; msk <<= 1) t = fmaxf(t, __shfl_xor(t, msk, 32));
      float mn = fmaxf(mrow[r], t);
      corr[r] = __expf(mrow[r] - mn);
      p0v[r] = __expf(x0 - mn);
      p1v[r] = __expf(x1 - mn);
      float rs = p0v[r] + p1v[r];
#pragma unroll
      for (int msk = 1; msk < 16; msk <<= 1) rs += __shfl_xor(rs, msk, 32);
      lrow[r] = lrow[r] * corr[r] + rs;
      mrow[r] = mn;
    }
#pragma unroll
    for (int r = 0; r < 8; r++) { acc0[r] *= corr[r]; acc1[r] *= corr[r]; }

    // transpose P through LDS (C layout -> A-fragment layout)
    {
      int row0 = (lane >> 4) * 8, cw = lane & 15;
#pragma unroll
      for (int r = 0; r < 8; r++) {
        lP[wv * 512 + (row0 + r) * 32 + cw]      = f2bf(p0v[r]);
        lP[wv * 512 + (row0 + r) * 32 + 16 + cw] = f2bf(p1v[r]);
      }
    }
    __syncthreads();

    Frag16 ap;
    {
      int c0 = (lane < 16) ? 0 : 16;
      const uint4* p = reinterpret_cast<const uint4*>(&lP[wv * 512 + (lane & 15) * 32]);
      ap.q[0] = p[c0 >> 3];
      ap.q[1] = p[(c0 >> 3) + 1];
    }
    // v fragments: B layout, N-col = d, K = m contiguous
    Frag16 bv0, bv1;
#if USE_TDM
    {
      const uint4* q0 = reinterpret_cast<const uint4*>(lV + (size_t)(lane & 15) * 32 + kb);
      bv0.q[0] = q0[0]; bv0.q[1] = q0[1];
      const uint4* q1 = reinterpret_cast<const uint4*>(lV + (size_t)(16 + (lane & 15)) * 32 + kb);
      bv1.q[0] = q1[0]; bv1.q[1] = q1[1];
    }
#else
    {
      int mm = mc + kb;
      const uint4* q0 = reinterpret_cast<const uint4*>(vh + (size_t)(lane & 15) * MT + mm);
      bv0.q[0] = q0[0]; bv0.q[1] = q0[1];
      const uint4* q1 = reinterpret_cast<const uint4*>(vh + (size_t)(16 + (lane & 15)) * MT + mm);
      bv1.q[0] = q1[0]; bv1.q[1] = q1[1];
    }
#endif
    acc0 = __builtin_amdgcn_wmma_f32_16x16x32_bf16(false, ap.v, false, bv0.v, (short)0, acc0, false, false);
    acc1 = __builtin_amdgcn_wmma_f32_16x16x32_bf16(false, ap.v, false, bv1.v, (short)0, acc1, false, false);
  }

  // finalize: ret = acc / l, then bn+gelu -> retT[n][l][VDEP] bf16 (channel-last)
  {
    int row0 = (lane >> 4) * 8, dcol = lane & 15;
#pragma unroll
    for (int r = 0; r < 8; r++) {
      float inv = 1.0f / lrow[r];
      int l = lb + row0 + r;
      {
        int c = h * DV + dcol;
        float g = geluf(acc0[r] * inv * s_ret[c] + b_ret[c]);
        retT[((size_t)n * LQ + l) * VDEP + c] = f2bf(g);
      }
      {
        int c = h * DV + 16 + dcol;
        float g = geluf(acc1[r] * inv * s_ret[c] + b_ret[c]);
        retT[((size_t)n * LQ + l) * VDEP + c] = f2bf(g);
      }
    }
  }
}

// ---------------- fp32 -> bf16 weight conversion ----------------
__global__ void cvt_bf16_kernel(const float* __restrict__ x, unsigned short* __restrict__ y, int count) {
  int i = blockIdx.x * 256 + threadIdx.x;
  if (i < count) y[i] = f2bf(x[i]);
}

// ---------------- host orchestration ----------------
extern "C" void kernel_launch(void* const* d_in, const int* in_sizes, int n_in,
                              void* d_out, int out_size, void* d_ws, size_t ws_size,
                              hipStream_t stream) {
  (void)in_sizes; (void)n_in; (void)out_size; (void)ws_size;

  const float* pixel = (const float*)d_in[0];
  const float* query = (const float*)d_in[1];

  char* base = (char*)d_ws;
  size_t off = 0;
  auto take = [&](size_t bytes) -> void* {
    off = (off + 255) & ~(size_t)255;
    void* p = base + off;
    off += bytes;
    return p;
  };

  unsigned short* wp1b  = (unsigned short*)take((size_t)BOT * CP * 2);
  unsigned short* wq1b  = (unsigned short*)take((size_t)BOT * CQ * 2);
  unsigned short* wqkvb = (unsigned short*)take((size_t)(2 * KD + VDEP) * BOT * 2);
  unsigned short* wpkvb = (unsigned short*)take((size_t)(KD + VDEP) * BOT * 2);
  unsigned short* wc3b  = (unsigned short*)take((size_t)CQ * VDEP * 2);
  unsigned short* wf1b  = (unsigned short*)take((size_t)FFN * CQ * 2);
  unsigned short* wf2b  = (unsigned short*)take((size_t)CQ * FFN * 2);
  unsigned short* Ps    = (unsigned short*)take((size_t)NB * HWD * BOT * 2);   // pixel_space CL
  unsigned short* Qs    = (unsigned short*)take((size_t)NB * LQ * BOT * 2);    // query_space CL
  unsigned short* qTb   = (unsigned short*)take((size_t)NB * NHD * LQ * 32 * 2);
  unsigned short* kTb   = (unsigned short*)take((size_t)NB * NHD * MT * 32 * 2);
  unsigned short* vBb   = (unsigned short*)take((size_t)NB * NHD * DV * MT * 2);
  unsigned short* retT  = (unsigned short*)take((size_t)NB * LQ * VDEP * 2);
  unsigned short* qfT   = (unsigned short*)take((size_t)NB * LQ * CQ * 2);
  float*          qf32  = (float*)take((size_t)NB * CQ * LQ * 4);
  unsigned short* f1T   = (unsigned short*)take((size_t)NB * LQ * FFN * 2);

  auto cvt = [&](const void* src, unsigned short* dst, int count) {
    cvt_bf16_kernel<<<(count + 255) / 256, 256, 0, stream>>>((const float*)src, dst, count);
  };
  cvt(d_in[5],  wp1b,  BOT * CP);
  cvt(d_in[2],  wq1b,  BOT * CQ);
  cvt(d_in[8],  wqkvb, (2 * KD + VDEP) * BOT);
  cvt(d_in[11], wpkvb, (KD + VDEP) * BOT);
  cvt(d_in[18], wc3b,  CQ * VDEP);
  cvt(d_in[21], wf1b,  FFN * CQ);
  cvt(d_in[24], wf2b,  CQ * FFN);

  // zero dk-pad regions (harness poisons ws with 0xAA)
  hipMemsetAsync(kTb, 0, (size_t)NB * NHD * MT * 32 * 2, stream);
  hipMemsetAsync(qTb, 0, (size_t)NB * NHD * LQ * 32 * 2, stream);

  // 1) pixel_space = gelu(bn(w_p1 x gelu(pixel)))   [fused gelu + transpose in B-load]
  {
    GemmArgs g{};
    g.A = wp1b; g.B = pixel;
    g.s = (const float*)d_in[6]; g.bias = (const float*)d_in[7];
    g.O = BOT; g.K = CP; g.M = HWD; g.bstride = (long long)CP * HWD;
    g.out_cl = Ps;
    gemm_wmma<B_CF_GELU, EPI_GELU_CL><<<dim3(HWD / 128, BOT / 128, NB), 256, 0, stream>>>(g);
  }
  // 2) query_space = gelu(bn(w_q1 x query))
  {
    GemmArgs g{};
    g.A = wq1b; g.B = query;
    g.s = (const float*)d_in[3]; g.bias = (const float*)d_in[4];
    g.O = BOT; g.K = CQ; g.M = LQ; g.bstride = (long long)CQ * LQ;
    g.out_cl = Qs;
    gemm_wmma<B_CF, EPI_GELU_CL><<<dim3(1, BOT / 128, NB), 256, 0, stream>>>(g);
  }
  // 3) pixel_kv = bn(w_pkv x pixel_space) -> kT (m+128), v (m+128)
  {
    GemmArgs g{};
    g.A = wpkvb; g.B = Ps;
    g.s = (const float*)d_in[12]; g.bias = (const float*)d_in[13];
    g.O = KD + VDEP; g.K = BOT; g.M = HWD; g.bstride = (long long)HWD * BOT;
    g.kT = kTb; g.vB = vBb; g.m_off = LQ;
    gemm_wmma<B_CL, EPI_PKV><<<dim3(HWD / 128, (KD + VDEP) / 128, NB), 256, 0, stream>>>(g);
  }
  // 4) query_qkv = bn(w_qkv x query_space) -> qT, kT (m), v (m)
  {
    GemmArgs g{};
    g.A = wqkvb; g.B = Qs;
    g.s = (const float*)d_in[9]; g.bias = (const float*)d_in[10];
    g.O = 2 * KD + VDEP; g.K = BOT; g.M = LQ; g.bstride = (long long)LQ * BOT;
    g.qT = qTb; g.kT = kTb; g.vB = vBb; g.m_off = 0;
    gemm_wmma<B_CL, EPI_QKV><<<dim3(1, (2 * KD + VDEP) / 128, NB), 256, 0, stream>>>(g);
  }
  // 5) attention (flash, WMMA both GEMMs, TDM-staged k/v) -> retT
  attn_wmma<<<NB * NHD, 256, 0, stream>>>(
      qTb, kTb, vBb,
      (const float*)d_in[14], (const float*)d_in[15],
      (const float*)d_in[16], (const float*)d_in[17], retT);
  // 6) qf = gelu(query + bn(w_c3 x ret))
  {
    GemmArgs g{};
    g.A = wc3b; g.B = retT;
    g.s = (const float*)d_in[19]; g.bias = (const float*)d_in[20];
    g.O = CQ; g.K = VDEP; g.M = LQ; g.bstride = (long long)LQ * VDEP;
    g.resid = query; g.out_cl = qfT; g.out_f32 = qf32;
    gemm_wmma<B_CL, EPI_QF><<<dim3(1, CQ / 128, NB), 256, 0, stream>>>(g);
  }
  // 7) ffn1 = gelu(bn(w_f1 x qf))
  {
    GemmArgs g{};
    g.A = wf1b; g.B = qfT;
    g.s = (const float*)d_in[22]; g.bias = (const float*)d_in[23];
    g.O = FFN; g.K = CQ; g.M = LQ; g.bstride = (long long)LQ * CQ;
    g.out_cl = f1T;
    gemm_wmma<B_CL, EPI_GELU_CL><<<dim3(1, FFN / 128, NB), 256, 0, stream>>>(g);
  }
  // 8) out = gelu(qf + bn(w_f2 x ffn1))
  {
    GemmArgs g{};
    g.A = wf2b; g.B = f1T;
    g.s = (const float*)d_in[25]; g.bias = (const float*)d_in[26];
    g.O = CQ; g.K = FFN; g.M = LQ; g.bstride = (long long)LQ * FFN;
    g.resid = qf32; g.out_f32 = (float*)d_out;
    gemm_wmma<B_CL, EPI_OUT><<<dim3(1, CQ / 128, NB), 256, 0, stream>>>(g);
  }
}
